// MultiHeadAttentionClassical_65481071398138
// MI455X (gfx1250) — matrix operand order repaired and checked by reading, not compiled, not run on hardware
//
#include <hip/hip_runtime.h>
#include <stdint.h>

typedef __attribute__((ext_vector_type(16))) __bf16 v16bf;
typedef __attribute__((ext_vector_type(8)))  __bf16 bf8v;
typedef __attribute__((ext_vector_type(8)))  float  v8f;
typedef __attribute__((ext_vector_type(8)))  float  f8v;

#define Bsz 4
#define Sq  2048
#define Emb 1024
#define Hn  16
#define Dk  64

// Native fp32 -> bf16 (RNE) via hardware cvt.
__device__ __forceinline__ unsigned short f2bf(float f) {
    union { __bf16 h; unsigned short u; } cv;
    cv.h = (__bf16)f;
    return cv.u;
}

__device__ __forceinline__ uint4 pack8(float4 f0, float4 f1) {
    f8v f = {f0.x, f0.y, f0.z, f0.w, f1.x, f1.y, f1.z, f1.w};
    union { bf8v b; uint4 u; } cv;
    cv.b = __builtin_convertvector(f, bf8v);
    return cv.u;
}

__device__ __forceinline__ v8f wmma_bf16(v16bf a, v16bf b, v8f c) {
    return __builtin_amdgcn_wmma_f32_16x16x32_bf16(
        /*neg_a=*/false, a, /*neg_b=*/false, b,
        /*c_mod=*/(short)0, c, /*reuse_a=*/false, /*reuse_b=*/false);
}

// A fragment (16x32 bf16): lane<16 -> row=lane, K={kb+0..7, kb+16..23}
//                          lane>=16 -> row=lane-16, K={kb+8..15, kb+24..31}
__device__ __forceinline__ v16bf fragA(const unsigned short* p, int rowElems,
                                       int rowBase, int kb) {
    int lane = threadIdx.x & 31;
    int half = lane >> 4;
    const unsigned short* q = p + (rowBase + (lane & 15)) * rowElems + kb + half * 8;
    v16bf a;
    ((uint4*)&a)[0] = *(const uint4*)q;
    ((uint4*)&a)[1] = *(const uint4*)(q + 16);
    return a;
}

// B fragment (32x16 bf16), LDS stored as [n][k]:
//   lane<16 -> n=lane, K={kb+0..7, kb+16..23} ; lane>=16 -> n=lane-16, K={kb+8..15, kb+24..31}
__device__ __forceinline__ v16bf fragB(const unsigned short* p, int rowElems,
                                       int rowBase, int kb) {
    int lane = threadIdx.x & 31;
    int half = lane >> 4;
    const unsigned short* q = p + (rowBase + (lane & 15)) * rowElems + kb + half * 8;
    v16bf b;
    ((uint4*)&b)[0] = *(const uint4*)q;
    ((uint4*)&b)[1] = *(const uint4*)(q + 16);
    return b;
}

// ---------------------------------------------------------------------------
// Kernel 1: QKV projection (one launch per MODE).  out = x @ W^T + b (bf16)
//   MODE 0 -> Q [b,h,s,dk],  MODE 1 -> K [b,h,s,dk],  MODE 2 -> V^T [b,h,dk,s]
// Block tile 128(M) x 128(N), K-step 32, software-pipelined global->LDS.
// ---------------------------------------------------------------------------
template <int MODE>
__global__ __launch_bounds__(256)
void qkv_proj_kernel(const float* __restrict__ x,
                     const float* __restrict__ w,
                     const float* __restrict__ bias,
                     unsigned short* __restrict__ dst) {
    __shared__ __align__(16) unsigned short sA[128 * 32];
    __shared__ __align__(16) unsigned short sB[128 * 32];

    const int mbase = blockIdx.x * 128;
    const int nbase = blockIdx.y * 128;
    const int t = threadIdx.x;
    const int wave = t >> 5, lane = t & 31;
    const int wm = wave & 3;      // M subtile (32 rows)
    const int wn = wave >> 2;     // N subtile (64 cols)
    const int lrow = t >> 1;          // 0..127
    const int lcol = (t & 1) * 16;    // 0 or 16

    const float* aSrc = x + (size_t)(mbase + lrow) * Emb + lcol;
    const float* bSrc = w + (size_t)(nbase + lrow) * Emb + lcol;

    float4 ar[4], br[4];
#pragma unroll
    for (int c = 0; c < 4; ++c) {
        ar[c] = *(const float4*)(aSrc + c * 4);
        br[c] = *(const float4*)(bSrc + c * 4);
    }

    v8f acc[2][4] = {};

    for (int kk = 0; kk < Emb; kk += 32) {
        *(uint4*)&sA[lrow * 32 + lcol]     = pack8(ar[0], ar[1]);
        *(uint4*)&sA[lrow * 32 + lcol + 8] = pack8(ar[2], ar[3]);
        *(uint4*)&sB[lrow * 32 + lcol]     = pack8(br[0], br[1]);
        *(uint4*)&sB[lrow * 32 + lcol + 8] = pack8(br[2], br[3]);
        __syncthreads();
        if (kk + 32 < Emb) {   // next K-tile into regs during compute
#pragma unroll
            for (int c = 0; c < 4; ++c) {
                ar[c] = *(const float4*)(aSrc + kk + 32 + c * 4);
                br[c] = *(const float4*)(bSrc + kk + 32 + c * 4);
            }
        }
        // preload all fragments, then burst 8 WMMAs
        v16bf af0 = fragA(sA, 32, wm * 32, 0);
        v16bf af1 = fragA(sA, 32, wm * 32 + 16, 0);
        v16bf bfr[4];
#pragma unroll
        for (int j = 0; j < 4; ++j) bfr[j] = fragB(sB, 32, wn * 64 + j * 16, 0);
#pragma unroll
        for (int j = 0; j < 4; ++j) {
            acc[0][j] = wmma_bf16(af0, bfr[j], acc[0][j]);
            acc[1][j] = wmma_bf16(af1, bfr[j], acc[1][j]);
        }
        __syncthreads();
    }

    const int half = lane >> 4;
#pragma unroll
    for (int i = 0; i < 2; ++i) {
        int mrowbase = mbase + wm * 32 + i * 16 + half * 8;
#pragma unroll
        for (int j = 0; j < 4; ++j) {
            int n = nbase + wn * 64 + j * 16 + (lane & 15);
            float bs = bias[n];
            int hh = n >> 6, dd = n & 63;
#pragma unroll
            for (int r = 0; r < 8; ++r) {
                int mrow = mrowbase + r;
                int bidx = mrow >> 11, ss = mrow & (Sq - 1);
                unsigned short val = f2bf(acc[i][j][r] + bs);
                if (MODE == 2)
                    dst[(size_t)((bidx * Hn + hh) * Dk + dd) * Sq + ss] = val;
                else
                    dst[(size_t)((bidx * Hn + hh) * Sq + ss) * Dk + dd] = val;
            }
        }
    }
}

// ---------------------------------------------------------------------------
// Kernel 2: flash attention. One block per (b, h, 128 q rows); 8 waves each
// owning 16 q rows. kv loop in steps of 64, register-pipelined K/V^T tiles.
// ---------------------------------------------------------------------------
__global__ __launch_bounds__(256)
void flash_attn_kernel(const unsigned short* __restrict__ qg,
                       const unsigned short* __restrict__ kg,
                       const unsigned short* __restrict__ vtg,
                       const int* __restrict__ mask,
                       unsigned short* __restrict__ og) {
    __shared__ __align__(16) unsigned short sQ[128 * 64];   // 16 KB
    __shared__ __align__(16) unsigned short sK[64 * 64];    //  8 KB
    __shared__ __align__(16) unsigned short sV[64 * 64];    //  8 KB (V^T tile)
    __shared__ __align__(16) unsigned short sP[8][16 * 64]; // 16 KB

    const int qb = blockIdx.x, hh = blockIdx.y, bb = blockIdx.z;
    const int bh = bb * Hn + hh;
    const int t = threadIdx.x;
    const int wave = t >> 5, lane = t & 31, half = lane >> 4;
    const int qbase = qb * 128;

    {   // load Q tile 128x64 bf16 (contiguous)
        const uint4* src = (const uint4*)(qg + (size_t)(bh * Sq + qbase) * Dk);
        uint4* dst = (uint4*)sQ;
#pragma unroll
        for (int i = 0; i < 4; ++i) dst[t + i * 256] = src[t + i * 256];
    }

    float m[8], l[8];
    v8f o[4] = {};
    for (int r = 0; r < 8; ++r) { m[r] = -3.0e38f; l[r] = 0.f; }

    const int vrow = t >> 2;          // 0..63 (dk row)
    const int vch  = (t & 3) * 16;    // 0,16,32,48 (kv chunk)

    const uint4* kSrc = (const uint4*)(kg + (size_t)bh * Sq * Dk);   // +kvb*8/u4
    const unsigned short* vSrc = vtg + (size_t)(bh * Dk + vrow) * Sq + vch;

    // register-stage kv tile 0
    uint4 kr0 = kSrc[t];
    uint4 kr1 = kSrc[t + 256];
    uint4 vr0 = *(const uint4*)(vSrc);
    uint4 vr1 = *(const uint4*)(vSrc + 8);

    __syncthreads();

    for (int kvb = 0; kvb < Sq; kvb += 64) {
        ((uint4*)sK)[t]       = kr0;
        ((uint4*)sK)[t + 256] = kr1;
        *(uint4*)&sV[vrow * 64 + vch]     = vr0;
        *(uint4*)&sV[vrow * 64 + vch + 8] = vr1;
        __syncthreads();

        if (kvb + 64 < Sq) {   // next tile into regs while we compute
            kr0 = kSrc[t + (kvb + 64) * 8];
            kr1 = kSrc[t + 256 + (kvb + 64) * 8];
            vr0 = *(const uint4*)(vSrc + kvb + 64);
            vr1 = *(const uint4*)(vSrc + kvb + 64 + 8);
            if (kvb + 128 < Sq) {   // warm L2 one tile further ahead
                __builtin_prefetch(kg + (size_t)(bh * Sq + kvb + 128) * Dk + t * 16, 0, 0);
                __builtin_prefetch(vtg + (size_t)(bh * Dk + (t & 63)) * Sq + kvb + 128
                                       + (t >> 6) * 16, 0, 0);
            }
        }

        // scores: S(16x64) = Qw(16x64) @ K^T  -- preload frags, burst WMMAs
        v8f s[4] = {};
        {
            v16bf a0 = fragA(sQ, 64, wave * 16, 0);
            v16bf bk[4];
#pragma unroll
            for (int nt = 0; nt < 4; ++nt) bk[nt] = fragB(sK, 64, nt * 16, 0);
#pragma unroll
            for (int nt = 0; nt < 4; ++nt) s[nt] = wmma_bf16(a0, bk[nt], s[nt]);
            v16bf a1 = fragA(sQ, 64, wave * 16, 32);
#pragma unroll
            for (int nt = 0; nt < 4; ++nt) bk[nt] = fragB(sK, 64, nt * 16, 32);
#pragma unroll
            for (int nt = 0; nt < 4; ++nt) s[nt] = wmma_bf16(a1, bk[nt], s[nt]);
        }

        // scale + padding mask (lane column fixed across all 8 rows)
        const float sc = 0.125f;  // 1/sqrt(DK)
#pragma unroll
        for (int nt = 0; nt < 4; ++nt) {
            bool mz = (mask[bb * Sq + kvb + nt * 16 + (lane & 15)] == 0);
#pragma unroll
            for (int r = 0; r < 8; ++r)
                s[nt][r] = mz ? -1e9f : s[nt][r] * sc;
        }

        // online softmax: row reductions across the 16 column-lanes
        float alpha[8];
#pragma unroll
        for (int r = 0; r < 8; ++r) {
            float mx = fmaxf(fmaxf(s[0][r], s[1][r]), fmaxf(s[2][r], s[3][r]));
            mx = fmaxf(mx, __shfl_xor(mx, 1, 32));
            mx = fmaxf(mx, __shfl_xor(mx, 2, 32));
            mx = fmaxf(mx, __shfl_xor(mx, 4, 32));
            mx = fmaxf(mx, __shfl_xor(mx, 8, 32));
            float mnew = fmaxf(m[r], mx);
            alpha[r] = __expf(m[r] - mnew);
            float ps = 0.f;
#pragma unroll
            for (int nt = 0; nt < 4; ++nt) {
                float p = __expf(s[nt][r] - mnew);
                s[nt][r] = p;
                ps += p;
            }
            ps += __shfl_xor(ps, 1, 32);
            ps += __shfl_xor(ps, 2, 32);
            ps += __shfl_xor(ps, 4, 32);
            ps += __shfl_xor(ps, 8, 32);
            l[r] = l[r] * alpha[r] + ps;
            m[r] = mnew;
        }
#pragma unroll
        for (int nt = 0; nt < 4; ++nt)
#pragma unroll
            for (int r = 0; r < 8; ++r) o[nt][r] *= alpha[r];

        // stage P (C-layout -> A-layout) via per-wave LDS
        unsigned short* pw = sP[wave];
#pragma unroll
        for (int r = 0; r < 8; ++r) {
            int row = r + 8 * half;
#pragma unroll
            for (int nt = 0; nt < 4; ++nt)
                pw[row * 64 + nt * 16 + (lane & 15)] = f2bf(s[nt][r]);
        }
        __builtin_amdgcn_wave_barrier();

        // PV: O(16x64) += P(16x64) @ V(64x64) -- preload frags, burst WMMAs
        {
            v16bf ap0 = fragA(pw, 64, 0, 0);
            v16bf bv[4];
#pragma unroll
            for (int nt = 0; nt < 4; ++nt) bv[nt] = fragB(sV, 64, nt * 16, 0);
#pragma unroll
            for (int nt = 0; nt < 4; ++nt) o[nt] = wmma_bf16(ap0, bv[nt], o[nt]);
            v16bf ap1 = fragA(pw, 64, 0, 32);
#pragma unroll
            for (int nt = 0; nt < 4; ++nt) bv[nt] = fragB(sV, 64, nt * 16, 32);
#pragma unroll
            for (int nt = 0; nt < 4; ++nt) o[nt] = wmma_bf16(ap1, bv[nt], o[nt]);
        }
        __syncthreads();
    }

    // normalize and write O [B*S][E] bf16
    const int mrow = qbase + wave * 16 + 8 * half;
#pragma unroll
    for (int r = 0; r < 8; ++r) {
        float inv = 1.0f / l[r];
        size_t gm = (size_t)(bb * Sq + mrow + r) * Emb + hh * Dk + (lane & 15);
#pragma unroll
        for (int nt = 0; nt < 4; ++nt)
            og[gm + nt * 16] = f2bf(o[nt][r] * inv);
    }
}

// ---------------------------------------------------------------------------
// Kernel 3: output projection  out = O @ wo^T + bo  (fp32 out)
// ---------------------------------------------------------------------------
__global__ __launch_bounds__(256)
void out_proj_kernel(const unsigned short* __restrict__ og,
                     const float* __restrict__ wo, const float* __restrict__ bo,
                     float* __restrict__ out) {
    __shared__ __align__(16) unsigned short sA[128 * 32];
    __shared__ __align__(16) unsigned short sB[128 * 32];

    const int mbase = blockIdx.x * 128;
    const int nbase = blockIdx.y * 128;
    const int t = threadIdx.x;
    const int wave = t >> 5, lane = t & 31;
    const int wm = wave & 3, wn = wave >> 2;
    const int lrow = t >> 1, lcol = (t & 1) * 16;

    const unsigned short* aSrc = og + (size_t)(mbase + lrow) * Emb + lcol;
    const float* bSrc = wo + (size_t)(nbase + lrow) * Emb + lcol;

    uint4 ar[2];
    float4 br[4];
    ar[0] = *(const uint4*)aSrc;
    ar[1] = *(const uint4*)(aSrc + 8);
#pragma unroll
    for (int c = 0; c < 4; ++c) br[c] = *(const float4*)(bSrc + c * 4);

    v8f acc[2][4] = {};

    for (int kk = 0; kk < Emb; kk += 32) {
        *(uint4*)&sA[lrow * 32 + lcol]     = ar[0];
        *(uint4*)&sA[lrow * 32 + lcol + 8] = ar[1];
        *(uint4*)&sB[lrow * 32 + lcol]     = pack8(br[0], br[1]);
        *(uint4*)&sB[lrow * 32 + lcol + 8] = pack8(br[2], br[3]);
        __syncthreads();
        if (kk + 32 < Emb) {
            ar[0] = *(const uint4*)(aSrc + kk + 32);
            ar[1] = *(const uint4*)(aSrc + kk + 32 + 8);
#pragma unroll
            for (int c = 0; c < 4; ++c)
                br[c] = *(const float4*)(bSrc + kk + 32 + c * 4);
        }
        v16bf af0 = fragA(sA, 32, wm * 32, 0);
        v16bf af1 = fragA(sA, 32, wm * 32 + 16, 0);
        v16bf bfr[4];
#pragma unroll
        for (int j = 0; j < 4; ++j) bfr[j] = fragB(sB, 32, wn * 64 + j * 16, 0);
#pragma unroll
        for (int j = 0; j < 4; ++j) {
            acc[0][j] = wmma_bf16(af0, bfr[j], acc[0][j]);
            acc[1][j] = wmma_bf16(af1, bfr[j], acc[1][j]);
        }
        __syncthreads();
    }

    const int half = lane >> 4;
#pragma unroll
    for (int i = 0; i < 2; ++i) {
        int mrow = mbase + wm * 32 + i * 16 + 8 * half;
#pragma unroll
        for (int j = 0; j < 4; ++j) {
            int n = nbase + wn * 64 + j * 16 + (lane & 15);
            float bs = bo[n];
#pragma unroll
            for (int r = 0; r < 8; ++r)
                out[(size_t)(mrow + r) * Emb + n] = acc[i][j][r] + bs;
        }
    }
}

// ---------------------------------------------------------------------------
extern "C" void kernel_launch(void* const* d_in, const int* in_sizes, int n_in,
                              void* d_out, int out_size, void* d_ws, size_t ws_size,
                              hipStream_t stream) {
    const float* x  = (const float*)d_in[0];
    const int*   mk = (const int*)d_in[1];
    const float* wq = (const float*)d_in[2];
    const float* bq = (const float*)d_in[3];
    const float* wk = (const float*)d_in[4];
    const float* bk = (const float*)d_in[5];
    const float* wv = (const float*)d_in[6];
    const float* bv = (const float*)d_in[7];
    const float* wo = (const float*)d_in[8];
    const float* bo = (const float*)d_in[9];
    float* out = (float*)d_out;

    const size_t elems = (size_t)Bsz * Sq * Emb;   // 8,388,608
    char* ws = (char*)d_ws;
    unsigned short* q  = (unsigned short*)ws;                 // 16 MB
    unsigned short* k  = (unsigned short*)(ws + 2 * elems);   // 16 MB
    unsigned short* vt = (unsigned short*)(ws + 4 * elems);   // 16 MB
    unsigned short* o  = (unsigned short*)(ws + 6 * elems);   // 16 MB

    dim3 blk(256);
    dim3 gemmGrid(64, 8, 1);
    qkv_proj_kernel<0><<<gemmGrid, blk, 0, stream>>>(x, wq, bq, q);
    qkv_proj_kernel<1><<<gemmGrid, blk, 0, stream>>>(x, wk, bk, k);
    qkv_proj_kernel<2><<<gemmGrid, blk, 0, stream>>>(x, wv, bv, vt);
    flash_attn_kernel<<<dim3(Sq / 128, Hn, Bsz), blk, 0, stream>>>(
        q, k, vt, mk, o);
    out_proj_kernel<<<gemmGrid, blk, 0, stream>>>(o, wo, bo, out);
}
